// CDGM_32959579030243
// MI455X (gfx1250) — matrix operand order repaired and checked by reading, not compiled, not run on hardware
//
#include <hip/hip_runtime.h>
#include <hip/hip_bf16.h>
#include <stdint.h>

// ---------------------------------------------------------------------------
// CDNA5 / gfx1250 dynamic-expert conv block.
// Dominant cost: per-sample 3x3 conv (38.7 GFLOP) -> bf16 WMMA implicit GEMM.
// Block tile 128(co) x 64(px) x 32(k), double-buffered LDS, 4 WMMA / wave / k.
// ---------------------------------------------------------------------------

typedef __bf16 bf16x8  __attribute__((ext_vector_type(8)));
typedef __bf16 bf16x16 __attribute__((ext_vector_type(16)));
typedef float  f32x8   __attribute__((ext_vector_type(8)));

__device__ __forceinline__ unsigned short f2bf(float f) {
  union { float f; uint32_t u; } v; v.f = f;
  uint32_t u = v.u;
  u += 0x7FFFu + ((u >> 16) & 1u);   // round-to-nearest-even
  return (unsigned short)(u >> 16);
}

// ---------------------------------------------------------------------------
// Kernel 1: attention  (one block per sample)
// ---------------------------------------------------------------------------
__global__ void attn_kernel(const float* __restrict__ sk,
                            const float* __restrict__ w1,
                            const float* __restrict__ w2,
                            float* __restrict__ att) {
  __shared__ float pooled[784];
  __shared__ float hdn[196];
  __shared__ float lg[16];
  const int b = blockIdx.x;
  const int t = threadIdx.x;
  const float* skb = sk + b * 56 * 56;
  for (int i = t; i < 784; i += 256) {
    int r = i / 28, c = i % 28;
    const float* p = skb + (2 * r) * 56 + 2 * c;
    pooled[i] = 0.25f * (p[0] + p[1] + p[56] + p[57]);
  }
  __syncthreads();
  if (t < 196) {
    float s = 0.f;
    const float* row = w1 + t * 784;
    for (int i = 0; i < 784; ++i) s += pooled[i] * row[i];
    hdn[t] = s > 0.f ? s : 0.f;
  }
  __syncthreads();
  if (t < 16) {
    float s = 0.f;
    const float* row = w2 + t * 196;
    for (int j = 0; j < 196; ++j) s += hdn[j] * row[j];
    lg[t] = s * (1.0f / 30.0f);
  }
  __syncthreads();
  if (t == 0) {
    float m = lg[0];
    for (int k = 1; k < 16; ++k) m = fmaxf(m, lg[k]);
    float e[16]; float sum = 0.f;
    for (int k = 0; k < 16; ++k) { e[k] = __expf(lg[k] - m); sum += e[k]; }
    float inv = 1.f / sum;
    for (int k = 0; k < 16; ++k) att[b * 16 + k] = e[k] * inv;
  }
}

// ---------------------------------------------------------------------------
// Kernel 2: x NCHW fp32 -> NHWC bf16
// ---------------------------------------------------------------------------
__global__ void xcvt_kernel(const float* __restrict__ xin,
                            unsigned short* __restrict__ xb) {
  int64_t idx = (int64_t)blockIdx.x * blockDim.x + threadIdx.x;
  const int64_t N = 8LL * 64 * 64 * 256;
  if (idx >= N) return;
  int c  = (int)(idx & 255);
  int xx = (int)((idx >> 8) & 63);
  int yy = (int)((idx >> 14) & 63);
  int b  = (int)(idx >> 20);
  xb[idx] = f2bf(xin[(((int64_t)b * 256 + c) << 12) + yy * 64 + xx]);
}

// ---------------------------------------------------------------------------
// Kernel 3: aggregate expert weights -> bf16 [b][tap][co][ci], plus agg bias
// ---------------------------------------------------------------------------
__global__ void agg_kernel(const float* __restrict__ att,
                           const float* __restrict__ weight,  // [K][co][ci][3][3]
                           const float* __restrict__ bias,    // [K][256]
                           unsigned short* __restrict__ aw,   // [b][9][co][ci] bf16
                           float* __restrict__ ab) {          // [b][256]
  const int64_t Nw = 8LL * 9 * 256 * 256;
  int64_t idx = (int64_t)blockIdx.x * blockDim.x + threadIdx.x;
  if (idx < Nw) {
    int ci  = (int)(idx & 255);
    int co  = (int)((idx >> 8) & 255);
    int tap = (int)((idx >> 16) % 9);
    int b   = (int)(idx / (9LL << 16));
    const int64_t wbase = (int64_t)co * 2304 + ci * 9 + tap;
    float s = 0.f;
#pragma unroll
    for (int k = 0; k < 16; ++k)
      s += att[b * 16 + k] * weight[(int64_t)k * 589824 + wbase];
    aw[idx] = f2bf(s);
  } else if (idx < Nw + 8 * 256) {
    int j = (int)(idx - Nw);
    int b = j >> 8, c = j & 255;
    float s = 0.f;
#pragma unroll
    for (int k = 0; k < 16; ++k) s += att[b * 16 + k] * bias[k * 256 + c];
    ab[j] = s;
  }
}

// ---------------------------------------------------------------------------
// Kernel 4: implicit-GEMM conv via v_wmma_f32_16x16x32_bf16
//   per sample: M=co(256), N=pixels(4096), K=ci*9(2304)
//   block tile: 128(co) x 64(one image row); 8 waves, tile grid 8x4;
//   wave = 2 M-tiles x 2 N-tiles = 4 accumulators; double-buffered LDS.
// ---------------------------------------------------------------------------
#define LDS_STRIDE 40   // 32 data halves + 8 pad halves (80B rows, 16B aligned)
#define KSTEPS 72

__global__ __launch_bounds__(256)
void conv_wmma_kernel(const unsigned short* __restrict__ xb,  // NHWC bf16
                      const unsigned short* __restrict__ aw,  // [b][9][co][ci] bf16
                      const float* __restrict__ ab,           // [b][256]
                      const float* __restrict__ xin,          // NCHW fp32 residual
                      float* __restrict__ out) {              // NCHW fp32
  __shared__ alignas(16) unsigned short Asm[2][128 * LDS_STRIDE];
  __shared__ alignas(16) unsigned short Bsm[2][64 * LDS_STRIDE];

  const int y      = blockIdx.x;          // image row == 64-px N tile
  const int coBase = blockIdx.y * 128;
  const int b      = blockIdx.z;
  const int t      = threadIdx.x;
  const int lane   = t & 31;
  const int w      = t >> 5;

  // tile-staging roles
  const int rowA0 = t >> 1;               // 0..127 : A row, segment pair
  const int segA0 = (t & 1) * 2;          // {0,2} -> two b128 segs per thread
  const int rowB  = t >> 2;               // 0..63
  const int segB  = t & 3;

  // wave tile assignment: 8 waves over 8(M) x 4(N) grid of 16x16 tiles
  const int mQ = w & 3;                   // M tile pair index (0..3) -> tiles 2mQ, 2mQ+1
  const int nH = w >> 2;                  // N tile pair index (0..1) -> tiles 2nH, 2nH+1

  f32x8 acc00 = {0.f,0.f,0.f,0.f,0.f,0.f,0.f,0.f};
  f32x8 acc01 = {0.f,0.f,0.f,0.f,0.f,0.f,0.f,0.f};
  f32x8 acc10 = {0.f,0.f,0.f,0.f,0.f,0.f,0.f,0.f};
  f32x8 acc11 = {0.f,0.f,0.f,0.f,0.f,0.f,0.f,0.f};

  // per-lane fragment offsets (halves), per ISA 7.12.2 layouts
  const int lane15 = lane & 15;
  const int hsel   = lane >> 4;
  const int aoff0 = ((mQ * 2 + 0) * 16 + lane15) * LDS_STRIDE + hsel * 8;
  const int aoff1 = ((mQ * 2 + 1) * 16 + lane15) * LDS_STRIDE + hsel * 8;
  const int boff0 = ((nH * 2 + 0) * 16 + lane15) * LDS_STRIDE + hsel * 8;
  const int boff1 = ((nH * 2 + 1) * 16 + lane15) * LDS_STRIDE + hsel * 8;

  uint4 ra0, ra1, rb;                      // staging registers

  // ---- global tile fetch for k-step kt into registers ----
  auto fetch = [&](int kt) {
    const int tap = kt >> 3;
    const int c0  = (kt & 7) << 5;
    const int kh  = tap / 3, kw = tap % 3;
    {
      const int64_t gA = ((((int64_t)b * 9 + tap) * 256) + coBase + rowA0) * 256 + c0;
      ra0 = *(const uint4*)(aw + gA + (segA0 + 0) * 8);
      ra1 = *(const uint4*)(aw + gA + (segA0 + 1) * 8);
    }
    {
      const int ys = y + kh - 1;
      const int xs = rowB + kw - 1;
      uint4 v = {0u, 0u, 0u, 0u};
      if ((unsigned)ys < 64u && (unsigned)xs < 64u) {
        const int64_t gB = ((((int64_t)b * 64 + ys) * 64) + xs) * 256 + c0 + segB * 8;
        v = *(const uint4*)(xb + gB);
      }
      rb = v;
    }
  };
  auto stage = [&](int buf) {
    *(uint4*)(&Asm[buf][rowA0 * LDS_STRIDE + (segA0 + 0) * 8]) = ra0;
    *(uint4*)(&Asm[buf][rowA0 * LDS_STRIDE + (segA0 + 1) * 8]) = ra1;
    *(uint4*)(&Bsm[buf][rowB  * LDS_STRIDE + segB * 8])        = rb;
  };

  fetch(0);
  stage(0);

  for (int kt = 0; kt < KSTEPS; ++kt) {
    const int cur = kt & 1;
    __syncthreads();                       // buf[cur] ready; buf[cur^1] free

    if (kt + 1 < KSTEPS) fetch(kt + 1);    // overlap global loads with WMMAs

    union FragU { bf16x16 v; struct { bf16x8 lo, hi; } p; } a0, a1, b0, b1;
    a0.p.lo = *(const bf16x8*)(&Asm[cur][aoff0]);
    a0.p.hi = *(const bf16x8*)(&Asm[cur][aoff0 + 16]);
    a1.p.lo = *(const bf16x8*)(&Asm[cur][aoff1]);
    a1.p.hi = *(const bf16x8*)(&Asm[cur][aoff1 + 16]);
    b0.p.lo = *(const bf16x8*)(&Bsm[cur][boff0]);
    b0.p.hi = *(const bf16x8*)(&Bsm[cur][boff0 + 16]);
    b1.p.lo = *(const bf16x8*)(&Bsm[cur][boff1]);
    b1.p.hi = *(const bf16x8*)(&Bsm[cur][boff1 + 16]);

    acc00 = __builtin_amdgcn_wmma_f32_16x16x32_bf16(false, a0.v, false, b0.v,
                                                    (short)0, acc00, false, false);
    acc01 = __builtin_amdgcn_wmma_f32_16x16x32_bf16(false, a0.v, false, b1.v,
                                                    (short)0, acc01, false, false);
    acc10 = __builtin_amdgcn_wmma_f32_16x16x32_bf16(false, a1.v, false, b0.v,
                                                    (short)0, acc10, false, false);
    acc11 = __builtin_amdgcn_wmma_f32_16x16x32_bf16(false, a1.v, false, b1.v,
                                                    (short)0, acc11, false, false);

    if (kt + 1 < KSTEPS) stage(cur ^ 1);   // write next tile after compute
  }

  // ---- epilogue: bias + residual, write NCHW fp32 ----
  const int x0 = (nH * 2 + 0) * 16 + lane15;
  const int x1 = x0 + 16;
  const int coA = coBase + (mQ * 2 + 0) * 16 + hsel * 8;  // D: VGPR r -> M=r (+8 hi lanes)
  const int coB = coBase + (mQ * 2 + 1) * 16 + hsel * 8;
#pragma unroll
  for (int r = 0; r < 8; ++r) {
    {
      const int co = coA + r;
      const float bb = ab[b * 256 + co];
      const int64_t base = (((int64_t)b * 256 + co) << 12) + y * 64;
      out[base + x0] = acc00[r] + bb + xin[base + x0];
      out[base + x1] = acc01[r] + bb + xin[base + x1];
    }
    {
      const int co = coB + r;
      const float bb = ab[b * 256 + co];
      const int64_t base = (((int64_t)b * 256 + co) << 12) + y * 64;
      out[base + x0] = acc10[r] + bb + xin[base + x0];
      out[base + x1] = acc11[r] + bb + xin[base + x1];
    }
  }
}

// ---------------------------------------------------------------------------
// Launch
// ---------------------------------------------------------------------------
extern "C" void kernel_launch(void* const* d_in, const int* in_sizes, int n_in,
                              void* d_out, int out_size, void* d_ws, size_t ws_size,
                              hipStream_t stream) {
  const float* x      = (const float*)d_in[0];  // [8,256,64,64]
  const float* sk     = (const float*)d_in[1];  // [8,1,56,56]
  const float* weight = (const float*)d_in[2];  // [16,256,256,3,3]
  const float* bias   = (const float*)d_in[3];  // [16,256]
  const float* w1     = (const float*)d_in[4];  // [196,784]
  const float* w2     = (const float*)d_in[5];  // [16,196]
  float* out = (float*)d_out;

  char* ws = (char*)d_ws;
  float* att          = (float*)(ws);                             // 512 B
  float* ab           = (float*)(ws + 4096);                      // 8 KB
  unsigned short* xb  = (unsigned short*)(ws + 65536);            // 16 MB NHWC bf16
  unsigned short* aw  = (unsigned short*)(ws + 65536 + 16777216); // 9 MB packed bf16

  attn_kernel<<<8, 256, 0, stream>>>(sk, w1, w2, att);

  {
    const int64_t N = 8LL * 64 * 64 * 256;
    xcvt_kernel<<<(int)((N + 255) / 256), 256, 0, stream>>>(x, xb);
  }
  {
    const int64_t N = 8LL * 9 * 256 * 256 + 8 * 256;
    agg_kernel<<<(int)((N + 255) / 256), 256, 0, stream>>>(att, weight, bias, aw, ab);
  }

  dim3 grid(64, 2, 8);  // (y-row, 128-co block, sample)
  conv_wmma_kernel<<<grid, 256, 0, stream>>>(xb, aw, ab, x, out);
}
